// Decoder_69595650064588
// MI455X (gfx1250) — matrix operand order repaired and checked by reading, not compiled, not run on hardware
//
#include <hip/hip_runtime.h>
#include <hip/hip_bf16.h>

static constexpr int Hn = 1024;   // hidden
static constexpr int Vn = 1024;   // vocab
static constexpr int Bn = 64;     // batch
static constexpr int Ln = 512;    // seq len

typedef __attribute__((ext_vector_type(16))) __bf16 v16bf;
typedef __attribute__((ext_vector_type(8)))  float  v8f;

union FragB16 { v16bf v; unsigned int d[8]; unsigned short u[16]; };

__device__ __forceinline__ unsigned short f32_to_bf16_rte(float f) {
  union { float f; unsigned int u; } x; x.f = f;
  unsigned int r = x.u + 0x7FFFu + ((x.u >> 16) & 1u);
  return (unsigned short)(r >> 16);
}

__device__ __forceinline__ float block_reduce_sum(float val, float* red) {
  int tid = threadIdx.x;
  red[tid] = val; __syncthreads();
  for (int s = 128; s > 0; s >>= 1) {
    if (tid < s) red[tid] += red[tid + s];
    __syncthreads();
  }
  float r = red[0]; __syncthreads();
  return r;
}

__device__ __forceinline__ float block_reduce_max(float val, float* red) {
  int tid = threadIdx.x;
  red[tid] = val; __syncthreads();
  for (int s = 128; s > 0; s >>= 1) {
    if (tid < s) red[tid] = fmaxf(red[tid], red[tid + s]);
    __syncthreads();
  }
  float r = red[0]; __syncthreads();
  return r;
}

// low 32 bits of a flat LDS pointer == LDS byte offset (aperture base low bits are 0)
__device__ __forceinline__ unsigned lds_off32(const void* p) {
  return (unsigned)(unsigned long long)p;
}

// ---------------------------------------------------------------------------
// Generic WMMA GEMM: C(64 x N) = A(64 x K, bf16 rowmajor) * W(N x K, bf16)^T + bias
// 256 threads = 8 wave32. Each wave: 16 cols, 64 rows (4 mtiles), K chunked by 32.
// Double-buffered LDS, staged with async global->LDS (ASYNCcnt) to overlap
// the next chunk's fill with the current chunk's WMMAs. One barrier per chunk.
// ---------------------------------------------------------------------------
#define SA_STRIDE 40   // shorts (80B: 16B aligned, conflict-free mod 64 banks)
#define SW_STRIDE 40

__global__ void __launch_bounds__(256)
gemm64_bf16_wmma(const unsigned short* __restrict__ A,
                 const unsigned short* __restrict__ W,
                 const float* __restrict__ bias,
                 float* __restrict__ C,
                 int K, int N)
{
  __shared__ unsigned short sA[2][64 * SA_STRIDE];
  __shared__ unsigned short sW[2][128 * SW_STRIDE];
  const int tid   = threadIdx.x;
  const int wave  = tid >> 5;
  const int lane  = tid & 31;
  const int lan15 = lane & 15;
  const int lhi   = lane >> 4;
  const int n_wg  = blockIdx.x * 128;

  // staging assignment: A = 4 threads/row x 16B; W = 2 threads/row x 2x16B
  const int ra = tid >> 2, ca = (tid & 3) * 8;
  const int rw = tid >> 1, cw = (tid & 1) * 16;
  const unsigned short* gA = A + (size_t)ra * K + ca;
  const unsigned short* gW = W + (size_t)(n_wg + rw) * K + cw;
  const unsigned ldsA[2] = { lds_off32(&sA[0][ra * SA_STRIDE + ca]),
                             lds_off32(&sA[1][ra * SA_STRIDE + ca]) };
  const unsigned ldsW[2] = { lds_off32(&sW[0][rw * SW_STRIDE + cw]),
                             lds_off32(&sW[1][rw * SW_STRIDE + cw]) };

  auto stage = [&](int bufi, int kc) {
    const unsigned short* ga = gA + kc;
    const unsigned short* gw = gW + kc;
    // async DMA: 16B global -> LDS copy per instruction, tracked with ASYNCcnt
    asm volatile("global_load_async_to_lds_b128 %0, %1, off"
                 :: "v"(ldsA[bufi]), "v"(ga) : "memory");
    asm volatile("global_load_async_to_lds_b128 %0, %1, off"
                 :: "v"(ldsW[bufi]), "v"(gw) : "memory");
    asm volatile("global_load_async_to_lds_b128 %0, %1, off offset:16"
                 :: "v"(ldsW[bufi]), "v"(gw) : "memory");
  };

  const v8f vzero = {0.f,0.f,0.f,0.f,0.f,0.f,0.f,0.f};
  v8f acc[4];
#pragma unroll
  for (int m = 0; m < 4; ++m) acc[m] = vzero;

  // prologue: fill buffer 0
  stage(0, 0);
  asm volatile("s_wait_asynccnt 0x0" ::: "memory");
  __syncthreads();

  int buf = 0;
  for (int kc = 0; kc < K; kc += 32) {
    const bool more = (kc + 32) < K;
    if (more) stage(buf ^ 1, kc + 32);   // overlap next fill with this chunk's math

    // B fragment (32x16): lanes 0-15 hold K=0..15, lanes 16-31 hold K=16..31
    FragB16 fb;
    {
      const int srow = wave * 16 + lan15;
      const unsigned int* p = (const unsigned int*)(&sW[buf][srow * SW_STRIDE]);
      const int kh = lhi * 8;
#pragma unroll
      for (int g = 0; g < 8; ++g) fb.d[g] = p[kh + g];
    }
#pragma unroll
    for (int m = 0; m < 4; ++m) {
      // A fragment (16x32): lanes 0-15 K=0..15, lanes 16-31 K=16..31
      FragB16 fa;
      const int row = m * 16 + lan15;
      const unsigned int* p = (const unsigned int*)(&sA[buf][row * SA_STRIDE]);
      const int kh = lhi * 8;
#pragma unroll
      for (int g = 0; g < 8; ++g) fa.d[g] = p[kh + g];
      acc[m] = __builtin_amdgcn_wmma_f32_16x16x32_bf16(
          false, fa.v, false, fb.v, (short)0, acc[m], false, false);
    }

    if (more) asm volatile("s_wait_asynccnt 0x0" ::: "memory");
    __syncthreads();
    buf ^= 1;
  }

  const int col = n_wg + wave * 16 + lan15;
  const float bval = bias[col];
#pragma unroll
  for (int m = 0; m < 4; ++m) {
#pragma unroll
    for (int r = 0; r < 8; ++r) {
      const int row = m * 16 + lhi * 8 + r;       // C layout: VGPR r, lanes16-31 = M+8
      C[(size_t)row * N + col] = acc[m][r] + bval;
    }
  }
}

// ---------------------------------------------------------------------------
// Precompute kernels
// ---------------------------------------------------------------------------
__global__ void __launch_bounds__(256)
prep_v_k(const float* __restrict__ Wa, const float* __restrict__ ba,
         const float* __restrict__ v, float* __restrict__ vh,
         float* __restrict__ ve, float* __restrict__ vb)
{
  const int i = blockIdx.x * 256 + threadIdx.x;   // 0..Hn-1
  float sh = 0.f, se = 0.f;
  for (int j = 0; j < Hn; ++j) {
    float vj = v[j];
    sh += Wa[(size_t)j * (2 * Hn) + i] * vj;
    se += Wa[(size_t)j * (2 * Hn) + Hn + i] * vj;
  }
  vh[i] = sh; ve[i] = se;
  if (blockIdx.x == 0) {
    __shared__ float red[256];
    float p = 0.f;
    for (int k = threadIdx.x; k < Hn; k += 256) p += ba[k] * v[k];
    p = block_reduce_sum(p, red);
    if (threadIdx.x == 0) vb[0] = p;
  }
}

__global__ void __launch_bounds__(256)
enc_score_k(const float* __restrict__ enc, const float* __restrict__ ve,
            float* __restrict__ score)
{
  __shared__ float red[256];
  const int l = blockIdx.x, b = blockIdx.y, tid = threadIdx.x;
  const float* e = enc + ((size_t)l * Bn + b) * Hn;
  float s = 0.f;
  for (int k = tid; k < Hn; k += 256) s += e[k] * ve[k];
  s = block_reduce_sum(s, red);
  if (tid == 0) score[(size_t)b * Ln + l] = s;
}

__global__ void __launch_bounds__(256)
convert_bf16_k(const float* __restrict__ src, unsigned short* __restrict__ dst, int n)
{
  int i = blockIdx.x * 256 + threadIdx.x;
  if (i < n) dst[i] = f32_to_bf16_rte(src[i]);
}

__global__ void __launch_bounds__(256)
init_state_k(const float* __restrict__ hidden, const float* __restrict__ dec0,
             float* __restrict__ hbuf, unsigned short* __restrict__ h_bf,
             unsigned short* __restrict__ xg, unsigned short* __restrict__ xo)
{
  const int i = blockIdx.x * 256 + threadIdx.x;   // Bn*Hn
  const int b = i >> 10, hh = i & (Hn - 1);
  float hv = hidden[i], dv = dec0[i];
  hbuf[i] = hv;
  h_bf[i] = f32_to_bf16_rte(hv);
  unsigned short db = f32_to_bf16_rte(dv);
  xg[(size_t)b * (2 * Hn) + hh]      = db;   // [dec | ctx]
  xo[(size_t)b * (3 * Hn) + Hn + hh] = db;   // [h_new | dec | ctx]
}

// ---------------------------------------------------------------------------
// Per-step kernels
// ---------------------------------------------------------------------------
__global__ void __launch_bounds__(256)
attn_softmax_k(const float* __restrict__ hbuf, const float* __restrict__ vh,
               const float* __restrict__ vbp, const float* __restrict__ score,
               float* __restrict__ a)
{
  __shared__ float red[256];
  const int b = blockIdx.x, tid = threadIdx.x;
  float p = 0.f;
  for (int k = tid; k < Hn; k += 256) p += hbuf[(size_t)b * Hn + k] * vh[k];
  float dot = block_reduce_sum(p, red) + vbp[0];
  float e0 = dot + score[(size_t)b * Ln + tid];
  float e1 = dot + score[(size_t)b * Ln + 256 + tid];
  float m = block_reduce_max(fmaxf(e0, e1), red);
  float x0 = __expf(e0 - m), x1 = __expf(e1 - m);
  float s = block_reduce_sum(x0 + x1, red);
  float inv = 1.f / s;
  a[(size_t)b * Ln + tid]       = x0 * inv;
  a[(size_t)b * Ln + 256 + tid] = x1 * inv;
}

__global__ void __launch_bounds__(256)
ctx_k(const float* __restrict__ a, const float* __restrict__ enc,
      float* __restrict__ ctx, unsigned short* __restrict__ xg,
      unsigned short* __restrict__ xo)
{
  __shared__ float sa[Ln];
  const int b = blockIdx.x, tid = threadIdx.x;
  const int h0 = blockIdx.y * 256 + tid;
  sa[tid]       = a[(size_t)b * Ln + tid];
  sa[tid + 256] = a[(size_t)b * Ln + 256 + tid];
  __syncthreads();
  const float* ep = enc + (size_t)b * Hn + h0;
  float s = 0.f;
#pragma unroll 4
  for (int l = 0; l < Ln; ++l) {
    if (l + 8 < Ln) __builtin_prefetch(ep + (size_t)(l + 8) * (Bn * Hn), 0, 0);
    s += sa[l] * ep[(size_t)l * (Bn * Hn)];
  }
  ctx[(size_t)b * Hn + h0] = s;
  unsigned short bs = f32_to_bf16_rte(s);
  xg[(size_t)b * (2 * Hn) + Hn + h0]     = bs;
  xo[(size_t)b * (3 * Hn) + 2 * Hn + h0] = bs;
}

__global__ void __launch_bounds__(256)
gru_gate_k(const float* __restrict__ gi, const float* __restrict__ gh,
           float* __restrict__ hbuf, unsigned short* __restrict__ h_bf,
           unsigned short* __restrict__ xo)
{
  const int i = blockIdx.x * 256 + threadIdx.x;
  const int b = i >> 10, hh = i & (Hn - 1);
  const size_t base = (size_t)b * (3 * Hn);
  float ir = gi[base + hh], iz = gi[base + Hn + hh], in_ = gi[base + 2 * Hn + hh];
  float hr = gh[base + hh], hz = gh[base + Hn + hh], hn = gh[base + 2 * Hn + hh];
  float r = 1.f / (1.f + __expf(-(ir + hr)));
  float z = 1.f / (1.f + __expf(-(iz + hz)));
  float n = tanhf(in_ + r * hn);
  float hold = hbuf[i];
  float hnew = (1.f - z) * n + z * hold;
  hbuf[i] = hnew;
  unsigned short hb = f32_to_bf16_rte(hnew);
  h_bf[i] = hb;
  xo[base + hh] = hb;
}

__global__ void __launch_bounds__(256)
softmax_dec_k(const float* __restrict__ logits,
              unsigned short* __restrict__ xg, unsigned short* __restrict__ xo,
              unsigned short* __restrict__ df)
{
  __shared__ float red[256];
  const int b = blockIdx.x, tid = threadIdx.x;
  const float* lp = logits + (size_t)b * Hn;
  float v0 = lp[tid], v1 = lp[tid + 256], v2 = lp[tid + 512], v3 = lp[tid + 768];
  float m = block_reduce_max(fmaxf(fmaxf(v0, v1), fmaxf(v2, v3)), red);
  float e0 = __expf(v0 - m), e1 = __expf(v1 - m), e2 = __expf(v2 - m), e3 = __expf(v3 - m);
  float s = block_reduce_sum(e0 + e1 + e2 + e3, red);
  float inv = 1.f / s;
  unsigned short d0 = f32_to_bf16_rte(e0 * inv), d1 = f32_to_bf16_rte(e1 * inv);
  unsigned short d2 = f32_to_bf16_rte(e2 * inv), d3 = f32_to_bf16_rte(e3 * inv);
  unsigned short* xgp = xg + (size_t)b * (2 * Hn);
  unsigned short* xop = xo + (size_t)b * (3 * Hn) + Hn;
  unsigned short* dfp = df + (size_t)b * Hn;
  xgp[tid] = d0; xgp[tid + 256] = d1; xgp[tid + 512] = d2; xgp[tid + 768] = d3;
  xop[tid] = d0; xop[tid + 256] = d1; xop[tid + 512] = d2; xop[tid + 768] = d3;
  dfp[tid] = d0; dfp[tid + 256] = d1; dfp[tid + 512] = d2; dfp[tid + 768] = d3;
}

__global__ void __launch_bounds__(256)
softmax_out_k(const float* __restrict__ logits, float* __restrict__ outp)
{
  __shared__ float red[256];
  const int b = blockIdx.x, tid = threadIdx.x;
  const float* lp = logits + (size_t)b * Vn;
  float v0 = lp[tid], v1 = lp[tid + 256], v2 = lp[tid + 512], v3 = lp[tid + 768];
  float m = block_reduce_max(fmaxf(fmaxf(v0, v1), fmaxf(v2, v3)), red);
  float e0 = __expf(v0 - m), e1 = __expf(v1 - m), e2 = __expf(v2 - m), e3 = __expf(v3 - m);
  float s = block_reduce_sum(e0 + e1 + e2 + e3, red);
  float inv = 1.f / s;
  float* op = outp + (size_t)b * Vn;
  op[tid] = e0 * inv; op[tid + 256] = e1 * inv;
  op[tid + 512] = e2 * inv; op[tid + 768] = e3 * inv;
}

// ---------------------------------------------------------------------------
extern "C" void kernel_launch(void* const* d_in, const int* in_sizes, int n_in,
                              void* d_out, int out_size, void* d_ws, size_t ws_size,
                              hipStream_t stream)
{
  const float* enc    = (const float*)d_in[0];
  const float* hidden = (const float*)d_in[1];
  const float* dec0   = (const float*)d_in[2];
  const float* Wa     = (const float*)d_in[3];
  const float* ba     = (const float*)d_in[4];
  const float* vvec   = (const float*)d_in[5];
  const float* W_ih   = (const float*)d_in[6];
  const float* b_ih   = (const float*)d_in[7];
  const float* W_hh   = (const float*)d_in[8];
  const float* b_hh   = (const float*)d_in[9];
  const float* Wo     = (const float*)d_in[10];
  const float* bo     = (const float*)d_in[11];
  const float* Wf     = (const float*)d_in[12];
  const float* bfv    = (const float*)d_in[13];
  float* out = (float*)d_out;

  char* ws = (char*)d_ws;
  size_t off = 0;
  auto alloc = [&](size_t bytes) -> void* {
    off = (off + 255) & ~(size_t)255;
    void* p = (void*)(ws + off);
    off += bytes;
    return p;
  };
  float* vh    = (float*)alloc((size_t)Hn * 4);
  float* ve    = (float*)alloc((size_t)Hn * 4);
  float* vb    = (float*)alloc(4);
  float* score = (float*)alloc((size_t)Bn * Ln * 4);
  float* aprob = (float*)alloc((size_t)Bn * Ln * 4);
  float* ctx   = (float*)alloc((size_t)Bn * Hn * 4);
  float* hbuf  = (float*)alloc((size_t)Bn * Hn * 4);
  float* gi    = (float*)alloc((size_t)Bn * 3 * Hn * 4);
  float* gh    = (float*)alloc((size_t)Bn * 3 * Hn * 4);
  float* olog  = (float*)alloc((size_t)Bn * Hn * 4);
  float* flog  = (float*)alloc((size_t)Bn * Vn * 4);
  unsigned short* h_bf   = (unsigned short*)alloc((size_t)Bn * Hn * 2);
  unsigned short* xg     = (unsigned short*)alloc((size_t)Bn * 2 * Hn * 2);
  unsigned short* xo     = (unsigned short*)alloc((size_t)Bn * 3 * Hn * 2);
  unsigned short* df     = (unsigned short*)alloc((size_t)Bn * Hn * 2);
  unsigned short* Wih_bf = (unsigned short*)alloc((size_t)3 * Hn * 2 * Hn * 2);
  unsigned short* Whh_bf = (unsigned short*)alloc((size_t)3 * Hn * Hn * 2);
  unsigned short* Wo_bf  = (unsigned short*)alloc((size_t)Hn * 3 * Hn * 2);
  unsigned short* Wf_bf  = (unsigned short*)alloc((size_t)Vn * Hn * 2);

  // ---- precompute ----
  prep_v_k<<<Hn / 256, 256, 0, stream>>>(Wa, ba, vvec, vh, ve, vb);
  {
    dim3 g(Ln, Bn);
    enc_score_k<<<g, 256, 0, stream>>>(enc, ve, score);
  }
  convert_bf16_k<<<(3 * Hn * 2 * Hn + 255) / 256, 256, 0, stream>>>(W_ih, Wih_bf, 3 * Hn * 2 * Hn);
  convert_bf16_k<<<(3 * Hn * Hn + 255) / 256, 256, 0, stream>>>(W_hh, Whh_bf, 3 * Hn * Hn);
  convert_bf16_k<<<(Hn * 3 * Hn + 255) / 256, 256, 0, stream>>>(Wo, Wo_bf, Hn * 3 * Hn);
  convert_bf16_k<<<(Vn * Hn + 255) / 256, 256, 0, stream>>>(Wf, Wf_bf, Vn * Hn);
  init_state_k<<<(Bn * Hn) / 256, 256, 0, stream>>>(hidden, dec0, hbuf, h_bf, xg, xo);

  // ---- sequential scan ----
  for (int t = 0; t < Ln; ++t) {
    attn_softmax_k<<<Bn, 256, 0, stream>>>(hbuf, vh, vb, score, aprob);
    {
      dim3 g(Bn, Hn / 256);
      ctx_k<<<g, 256, 0, stream>>>(aprob, enc, ctx, xg, xo);
    }
    gemm64_bf16_wmma<<<(3 * Hn) / 128, 256, 0, stream>>>(xg,   Wih_bf, b_ih, gi, 2 * Hn, 3 * Hn);
    gemm64_bf16_wmma<<<(3 * Hn) / 128, 256, 0, stream>>>(h_bf, Whh_bf, b_hh, gh, Hn,     3 * Hn);
    gru_gate_k<<<(Bn * Hn) / 256, 256, 0, stream>>>(gi, gh, hbuf, h_bf, xo);
    gemm64_bf16_wmma<<<Hn / 128, 256, 0, stream>>>(xo, Wo_bf, bo,  olog, 3 * Hn, Hn);
    softmax_dec_k<<<Bn, 256, 0, stream>>>(olog, xg, xo, df);
    gemm64_bf16_wmma<<<Vn / 128, 256, 0, stream>>>(df, Wf_bf, bfv, flog, Hn, Vn);
    softmax_out_k<<<Bn, 256, 0, stream>>>(flog, out + (size_t)t * Bn * Vn);
  }
}
